// ReconstructionDecoder_36421322670264
// MI455X (gfx1250) — matrix operand order repaired and checked by reading, not compiled, not run on hardware
//
#include <hip/hip_runtime.h>
#include <hip/hip_bf16.h>

typedef __attribute__((ext_vector_type(16))) _Float16 v16h;
typedef __attribute__((ext_vector_type(8)))  float    v8f;

#define D_DIM   256
#define E_DIM   512
#define TF_TOT  16384          // T*F
#define G_TILES 4              // (t,f) tiles per workgroup
#define EPS     1e-5f

// ---------------------------------------------------------------------------
// Pre-kernel: convert W1 [256 x 512] f32 -> f16, stored as per-lane WMMA
// B-fragments.  Fragment id = nt*8 + kk  (nt = N-tile 0..31, kk = K-step 0..7).
// B packing used here (contiguous k order, matched by A): lane L holds column n = L%16; halves i=0..15
// hold K = kk*32 + (L>=16 ? 16 : 0) + i.  Each lane's 16 halves are stored
// contiguously (32 B) so the main kernel loads one v16h per fragment.
// ---------------------------------------------------------------------------
__global__ void convert_w1_kernel(const float* __restrict__ W1,
                                  _Float16* __restrict__ w1h) {
    const int frag = blockIdx.x;          // 0..255
    const int nt   = frag >> 3;
    const int kk   = frag & 7;
    const int lane = threadIdx.x;         // 0..31
    const int lo   = lane & 15;
    const int hi   = lane >> 4;

    v16h v;
#pragma unroll
    for (int i = 0; i < 16; ++i) {
        const int K = kk * 32 + hi * 16 + i;       // row of W1 (d index)
        const int n = nt * 16 + lo;                // col of W1 (e index)
        v[i] = (_Float16)W1[K * E_DIM + n];
    }
    *(v16h*)(w1h + ((size_t)frag * 32 + lane) * 16) = v;
}

// ---------------------------------------------------------------------------
// Main kernel: one workgroup = G_TILES consecutive (t,f) tiles.
//   Stage 1: build A = f16( LayerNorm(x + t_emb + f_emb) )  [4 x 16 x 256] in LDS
//   Stage 2: WMMA GEMM1 (K=256, N=512) + bias + ReLU, fused GEMM2 (512->2)
//   Stage 3: cross-wave reduction, add b2, store [16 rows x 2] per tile
// ---------------------------------------------------------------------------
__global__ __launch_bounds__(256)
void recon_decoder_kernel(const float* __restrict__ x,      // [16 x 256]
                          const float* __restrict__ t_emb,  // [128 x 256]
                          const float* __restrict__ f_emb,  // [128 x 256]
                          const float* __restrict__ ln_g,
                          const float* __restrict__ ln_b,
                          const _Float16* __restrict__ w1h, // packed B frags
                          const float* __restrict__ b1,     // [512]
                          const float* __restrict__ W2,     // [512 x 2]
                          const float* __restrict__ b2,     // [2]
                          float* __restrict__ out) {        // [16 x 16384 x 2]
    __shared__ _Float16 A_lds[G_TILES * 16 * D_DIM];        // 32 KB
    __shared__ float    out_part[8 * G_TILES * 16 * 2];     // 4 KB

    const int tid  = threadIdx.x;
    const int w    = tid >> 5;     // wave 0..7
    const int lane = tid & 31;
    const int lo   = lane & 15;
    const int hi   = lane >> 4;

    // ---------------- Stage 1: LayerNorm rows into LDS (f16) ---------------
    {
        const int m   = w & 3;                 // which (t,f) tile
        const int row = ((w >> 2) << 3) + (lane & 7);   // bn row 0..15
        const int q   = lane >> 3 & 3;         // quarter of D: 64 elems
        const int tf  = blockIdx.x * G_TILES + m;
        const int t   = tf >> 7;
        const int f   = tf & 127;

        const float4* x4 = (const float4*)(x     + row * D_DIM + q * 64);
        const float4* t4 = (const float4*)(t_emb + t   * D_DIM + q * 64);
        const float4* f4 = (const float4*)(f_emb + f   * D_DIM + q * 64);

        float s = 0.f, s2 = 0.f;
#pragma unroll 4
        for (int i = 0; i < 16; ++i) {
            float4 xv = x4[i], tv = t4[i], fv = f4[i];
            float h0 = xv.x + tv.x + fv.x;
            float h1 = xv.y + tv.y + fv.y;
            float h2 = xv.z + tv.z + fv.z;
            float h3 = xv.w + tv.w + fv.w;
            s  += h0 + h1 + h2 + h3;
            s2 += h0*h0 + h1*h1 + h2*h2 + h3*h3;
        }
        // reduce over the 4 lanes (q = lane bits 3,4) sharing this row
        s  += __shfl_xor(s,  8);  s  += __shfl_xor(s,  16);
        s2 += __shfl_xor(s2, 8);  s2 += __shfl_xor(s2, 16);

        const float mean = s * (1.f / 256.f);
        const float var  = s2 * (1.f / 256.f) - mean * mean;
        const float rstd = rsqrtf(var + EPS);

        const float4* g4 = (const float4*)(ln_g + q * 64);
        const float4* c4 = (const float4*)(ln_b + q * 64);
        const int hbase = (m * 16 + row) * D_DIM + q * 64;
#pragma unroll 4
        for (int i = 0; i < 16; ++i) {
            float4 xv = x4[i], tv = t4[i], fv = f4[i];
            float4 gv = g4[i], cv = c4[i];
            union { _Float16 h[4]; uint2 u; } pk;
            pk.h[0] = (_Float16)(((xv.x + tv.x + fv.x) - mean) * rstd * gv.x + cv.x);
            pk.h[1] = (_Float16)(((xv.y + tv.y + fv.y) - mean) * rstd * gv.y + cv.y);
            pk.h[2] = (_Float16)(((xv.z + tv.z + fv.z) - mean) * rstd * gv.z + cv.z);
            pk.h[3] = (_Float16)(((xv.w + tv.w + fv.w) - mean) * rstd * gv.w + cv.w);
            *(uint2*)&A_lds[hbase + i * 4] = pk.u;
        }
    }
    __syncthreads();

    // ---------------- Stage 2: WMMA GEMM1 + ReLU + fused GEMM2 --------------
    // Wave w owns N-tiles nt = w*4 .. w*4+3 for ALL G_TILES M-tiles.
    float po[G_TILES][8][2];
#pragma unroll
    for (int m = 0; m < G_TILES; ++m)
#pragma unroll
        for (int r = 0; r < 8; ++r) { po[m][r][0] = 0.f; po[m][r][1] = 0.f; }

    for (int j = 0; j < 4; ++j) {
        const int nt = w * 4 + j;
        const int e  = nt * 16 + lo;      // this lane's output column of GEMM1
        const float b1v = b1[e];
        const float w20 = W2[e * 2 + 0];
        const float w21 = W2[e * 2 + 1];

        v8f acc[G_TILES];
#pragma unroll
        for (int m = 0; m < G_TILES; ++m) acc[m] = (v8f){0,0,0,0,0,0,0,0};

        for (int kk = 0; kk < 8; ++kk) {
            // B fragment: one contiguous 32-byte load per lane (L2-resident)
            const v16h bfrag =
                *(const v16h*)(w1h + (((size_t)nt * 8 + kk) * 32 + lane) * 16);
#pragma unroll
            for (int m = 0; m < G_TILES; ++m) {
                // A fragment, K = kk*32 + hi*16 + i (matches B packing): two 16-byte LDS reads.
                const int rb = (m * 16 + lo) * D_DIM + kk * 32;
                union { uint4 u[2]; v16h v; } af;
                af.u[0] = *(const uint4*)&A_lds[rb + hi * 16];
                af.u[1] = *(const uint4*)&A_lds[rb + hi * 16 + 8];
                acc[m] = __builtin_amdgcn_wmma_f32_16x16x32_f16(
                    false, af.v, false, bfrag, (short)0, acc[m], false, false);
            }
        }
        // bias + ReLU + multiply into the 512->2 projection (per-lane partials)
#pragma unroll
        for (int m = 0; m < G_TILES; ++m)
#pragma unroll
            for (int r = 0; r < 8; ++r) {
                float v = acc[m][r] + b1v;
                v = fmaxf(v, 0.f);
                po[m][r][0] += v * w20;
                po[m][r][1] += v * w21;
            }
    }

    // Reduce partials over the 16 lanes of each half (sum over e columns).
#pragma unroll
    for (int m = 0; m < G_TILES; ++m)
#pragma unroll
        for (int r = 0; r < 8; ++r)
#pragma unroll
            for (int o = 0; o < 2; ++o) {
                float v = po[m][r][o];
                v += __shfl_xor(v, 1);
                v += __shfl_xor(v, 2);
                v += __shfl_xor(v, 4);
                v += __shfl_xor(v, 8);
                if (lo == 0) {
                    const int row = hi * 8 + r;   // lane 0 -> rows 0..7, lane 16 -> 8..15
                    out_part[((w * G_TILES + m) * 16 + row) * 2 + o] = v;
                }
            }
    __syncthreads();

    // ---------------- Stage 3: cross-wave reduction + store -----------------
    if (tid < G_TILES * 16 * 2) {
        const int o   = tid & 1;
        const int row = (tid >> 1) & 15;
        const int m   = tid >> 5;
        float s = b2[o];
#pragma unroll
        for (int ww = 0; ww < 8; ++ww)
            s += out_part[((ww * G_TILES + m) * 16 + row) * 2 + o];
        const int tf = blockIdx.x * G_TILES + m;
        out[((size_t)row * TF_TOT + tf) * 2 + o] = s;
    }
}

// ---------------------------------------------------------------------------
extern "C" void kernel_launch(void* const* d_in, const int* in_sizes, int n_in,
                              void* d_out, int out_size, void* d_ws, size_t ws_size,
                              hipStream_t stream) {
    const float* x     = (const float*)d_in[0];
    const float* t_emb = (const float*)d_in[1];
    const float* f_emb = (const float*)d_in[2];
    const float* ln_g  = (const float*)d_in[3];
    const float* ln_b  = (const float*)d_in[4];
    const float* W1    = (const float*)d_in[5];
    const float* b1    = (const float*)d_in[6];
    const float* W2    = (const float*)d_in[7];
    const float* b2    = (const float*)d_in[8];
    float* out = (float*)d_out;

    _Float16* w1h = (_Float16*)d_ws;   // 256 KB packed f16 W1 fragments

    convert_w1_kernel<<<256, 32, 0, stream>>>(W1, w1h);
    recon_decoder_kernel<<<TF_TOT / G_TILES, 256, 0, stream>>>(
        x, t_emb, f_emb, ln_g, ln_b, w1h, b1, W2, b2, out);
}
